// WeakREChead_47004122087814
// MI455X (gfx1250) — compile-verified, matrix operand order checked by reading, not run on hardware
//
#include <hip/hip_runtime.h>
#include <hip/hip_bf16.h>
#include <math.h>

// Problem constants (from reference setup_inputs)
#define BB 128   // batch
#define VV 784   // vision regions
#define DD 512   // embed dim

typedef __attribute__((ext_vector_type(16))) _Float16 v16h;
typedef __attribute__((ext_vector_type(8)))  _Float16 v8h;
typedef __attribute__((ext_vector_type(8)))  float    v8f;
typedef __attribute__((ext_vector_type(4)))  float    v4f;

union AFrag { v16h v; v8h h8[2]; };
union BFrag { v16h v; _Float16 h[16]; };
union CFrag { v8f  v; float f[8]; };

// ---------------------------------------------------------------------------
// Kernel 0: convert lan_fs [B][3][D] f32 -> lan16 [c][b][d] f16 (hot in L2)
// ---------------------------------------------------------------------------
__global__ __launch_bounds__(256) void cvt_lan_kernel(
    const float* __restrict__ lan, _Float16* __restrict__ lan16)
{
  int idx = blockIdx.x * blockDim.x + threadIdx.x;
  if (idx < 3 * BB * DD) {
    int d = idx % DD;
    int b = (idx / DD) % BB;
    int c = idx / (DD * BB);
    lan16[idx] = (_Float16)lan[((size_t)b * 3 + c) * DD + d];
  }
}

// ---------------------------------------------------------------------------
// Kernel A: fused  sim[b,a,v] = lan_s[b].vis[a,v]  +  top-2 over v.
// grid = (case, a)  -- case is the FAST grid dimension so the 3 blocks that
// stream the same 1.6 MB vis[a] slice are launch-adjacent and share L2:
// vis HBM traffic ~205 MB (the once-through floor) instead of 3x that.
// 8 waves; wave w owns b-rows [16w,16w+16).
// M=b (A = lan f16), N=v (B = vis f32 -> f16 in regs), K=d, v_wmma f32 acc.
// Per ISA 7.12.2: 16-bit A 16x32 -> lane l holds row l%16, halves 0..7 = K
// base..base+7 (base = (l<16?0:8)), halves 8..15 = K 16+base.. ; B mirrored.
// C 16x16 f32: lane l col l%16, VGPR r row r + (l<16?0:8).
// ---------------------------------------------------------------------------
__global__ __launch_bounds__(256) void sim_topk_kernel(
    const float* __restrict__ vis, const _Float16* __restrict__ lan16,
    float* __restrict__ max0, float* __restrict__ max1, int* __restrict__ diagidx)
{
  const int c    = blockIdx.x;   // fast dim: 3 cases share L2-resident vis[a]
  const int a    = blockIdx.y;
  const int tid  = threadIdx.x;
  const int wave = tid >> 5;
  const int lane = tid & 31;
  const int lh   = lane & 15;
  const int hi   = lane >> 4;
  const int kbase = hi << 3;

  const _Float16* lanRow = lan16 + ((size_t)c * BB + wave * 16 + lh) * DD;
  const float*    visA   = vis + (size_t)a * VV * DD;

  float b1v[8], b2v[8];
  int   b1i[8];
#pragma unroll
  for (int r = 0; r < 8; ++r) { b1v[r] = -3.0e38f; b2v[r] = -3.0e38f; b1i[r] = 0; }

  // 49 v-tiles of 16 = 784, processed as 7 chunks of 7 tiles (112 rows each)
  for (int chunk = 0; chunk < 7; ++chunk) {
    const int vbase = chunk * 112;

    // Prefetch next chunk's 112 vis rows (224 KB) into the cache hierarchy
    // (global_prefetch_b8; speculative, no counters, capture-safe).
    if (chunk < 6) {
      const char* nxt = (const char*)(visA + (size_t)(vbase + 112) * DD);
#pragma unroll
      for (int i = 0; i < 4; ++i)
        __builtin_prefetch(nxt + (size_t)tid * 256 + (size_t)i * 65536, 0, 0);
    }

    v8f acc[7];
#pragma unroll
    for (int t = 0; t < 7; ++t) acc[t] = (v8f){0.f,0.f,0.f,0.f,0.f,0.f,0.f,0.f};

    for (int kb = 0; kb < 16; ++kb) {          // K = 512 / 32
      const int d0 = (kb << 5) + kbase;
      AFrag af;
      af.h8[0] = *(const v8h*)(lanRow + d0);        // K = base..base+7
      af.h8[1] = *(const v8h*)(lanRow + d0 + 16);   // K = 16+base..
#pragma unroll
      for (int t = 0; t < 7; ++t) {
        const float* vrow = visA + (size_t)(vbase + t * 16 + lh) * DD;
        v4f f0 = *(const v4f*)(vrow + d0);
        v4f f1 = *(const v4f*)(vrow + d0 + 4);
        v4f f2 = *(const v4f*)(vrow + d0 + 16);
        v4f f3 = *(const v4f*)(vrow + d0 + 20);
        BFrag bf;
#pragma unroll
        for (int i = 0; i < 4; ++i) {
          bf.h[i]      = (_Float16)f0[i];
          bf.h[4 + i]  = (_Float16)f1[i];
          bf.h[8 + i]  = (_Float16)f2[i];
          bf.h[12 + i] = (_Float16)f3[i];
        }
        acc[t] = __builtin_amdgcn_wmma_f32_16x16x32_f16(
            false, af.v, false, bf.v, (short)0, acc[t], false, false);
      }
    }

    // fold this chunk into the running per-lane top-2
#pragma unroll
    for (int t = 0; t < 7; ++t) {
      CFrag cf; cf.v = acc[t];
      const int vcol = vbase + t * 16 + lh;
#pragma unroll
      for (int r = 0; r < 8; ++r) {
        float val = cf.f[r];
        if (val > b1v[r]) { b2v[r] = b1v[r]; b1v[r] = val; b1i[r] = vcol; }
        else if (val > b2v[r]) { b2v[r] = val; }
      }
    }
  }

  // cross-lane merge: each b-row's 16 candidate pairs -> final top-2
  __shared__ float s1[BB][16];
  __shared__ float s2[BB][16];
  __shared__ int   si[BB][16];
#pragma unroll
  for (int r = 0; r < 8; ++r) {
    int brow = wave * 16 + hi * 8 + r;   // C-layout row for VGPR r
    s1[brow][lh] = b1v[r];
    s2[brow][lh] = b2v[r];
    si[brow][lh] = b1i[r];
  }
  __syncthreads();
  if (tid < BB) {
    const int b = tid;
    float t1 = -3.0e38f, t2 = -3.0e38f;
    int ti = 0;
#pragma unroll
    for (int j = 0; j < 16; ++j) {
      float c1 = s1[b][j], c2 = s2[b][j];
      if (c1 > t1) { t2 = (t1 > c2) ? t1 : c2; t1 = c1; ti = si[b][j]; }
      else         { t2 = (c1 > t2) ? c1 : t2; }
    }
    size_t idx = ((size_t)c * BB + b) * BB + a;
    max0[idx] = t1;
    max1[idx] = t2;
    if (b == a) diagidx[c * BB + a] = ti;
  }
}

// ---------------------------------------------------------------------------
// Kernel B: selected = vis[i, diag_idx[i]]; the three tiny B x B f32 GEMMs.
// grid = (i, case), 128 threads = column j.
// ---------------------------------------------------------------------------
__global__ __launch_bounds__(128) void adj_kernel(
    const float* __restrict__ vis, const float* __restrict__ lan,
    const int* __restrict__ diagidx,
    float* __restrict__ lan_adj, float* __restrict__ img_adj,
    float* __restrict__ logits)
{
  const int i = blockIdx.x, c = blockIdx.y, j = threadIdx.x;
  __shared__ float seli[DD];
  __shared__ float lani[DD];
  const float* selRowI = vis + ((size_t)i * VV + diagidx[c * BB + i]) * DD;
  const float* lanRowI = lan + ((size_t)i * 3 + c) * DD;
  for (int d = j; d < DD; d += 128) { seli[d] = selRowI[d]; lani[d] = lanRowI[d]; }
  __syncthreads();

  const float* selRowJ = vis + ((size_t)j * VV + diagidx[c * BB + j]) * DD;
  const float* lanRowJ = lan + ((size_t)j * 3 + c) * DD;
  float sII = 0.f, sIL = 0.f, sLL = 0.f;
  for (int d = 0; d < DD; ++d) {
    float vj = selRowJ[d], lj = lanRowJ[d];
    sII += seli[d] * vj;
    sIL += seli[d] * lj;
    sLL += lani[d] * lj;
  }
  size_t idx = ((size_t)c * BB + i) * BB + j;
  img_adj[idx] = sII;   // selected @ selected.T
  logits[idx]  = sIL;   // selected @ lan_s.T
  lan_adj[idx] = sLL;   // lan_s @ lan_s.T
}

// ---------------------------------------------------------------------------
// Deterministic block reductions (128 threads), broadcast result
// ---------------------------------------------------------------------------
__device__ __forceinline__ float bmax128(float v, float* red) {
  int j = threadIdx.x;
  red[j] = v; __syncthreads();
  for (int s = 64; s > 0; s >>= 1) {
    if (j < s) red[j] = fmaxf(red[j], red[j + s]);
    __syncthreads();
  }
  float r = red[0]; __syncthreads();
  return r;
}
__device__ __forceinline__ float bsum128(float v, float* red) {
  int j = threadIdx.x;
  red[j] = v; __syncthreads();
  for (int s = 64; s > 0; s >>= 1) {
    if (j < s) red[j] = red[j] + red[j + s];
    __syncthreads();
  }
  float r = red[0]; __syncthreads();
  return r;
}

// ---------------------------------------------------------------------------
// Kernel C1: per (case,row): CE lse over 255-entry concat row, score row
// (softmax of max0), and sum |softmax((img+lan)/2) - softmax(logits)|.
// ---------------------------------------------------------------------------
__global__ __launch_bounds__(128) void row_loss_kernel(
    const float* __restrict__ max0, const float* __restrict__ max1,
    const float* __restrict__ lan_adj, const float* __restrict__ img_adj,
    const float* __restrict__ logits,
    float* __restrict__ score, float* __restrict__ ce_row,
    float* __restrict__ fns_row)
{
  __shared__ float red[128];
  const int b = blockIdx.x, c = blockIdx.y, j = threadIdx.x;
  const size_t rowOff = ((size_t)c * BB + b) * BB;
  float x0 = max0[rowOff + j];
  float x1 = max1[rowOff + j];
  const bool off = (j != b);

  // CE over new_logits = [max0 row (B)] ++ [max1 row without col b (B-1)]
  float lmax   = off ? fmaxf(x0, x1) : x0;
  float rowmax = bmax128(lmax, red);
  float es     = expf(x0 - rowmax) + (off ? expf(x1 - rowmax) : 0.0f);
  float lse    = rowmax + logf(bsum128(es, red));
  if (j == 0) ce_row[c * BB + b] = -(max0[rowOff + b] - lse);

  // score = softmax(max0 row)
  float m0 = bmax128(x0, red);
  float e0 = expf(x0 - m0);
  float s0 = bsum128(e0, red);
  score[rowOff + j] = e0 / s0;

  // loss_fns row: |softmax((img+lan)/2) - softmax(logits)|
  float q  = 0.5f * (img_adj[rowOff + j] + lan_adj[rowOff + j]);
  float lg = logits[rowOff + j];
  float mq = bmax128(q, red);
  float eq = expf(q - mq);
  float sq = bsum128(eq, red);
  float ml = bmax128(lg, red);
  float el = expf(lg - ml);
  float sl = bsum128(el, red);
  float sd = bsum128(fabsf(eq / sq - el / sl), red);
  if (j == 0) fns_row[c * BB + b] = sd;
}

// ---------------------------------------------------------------------------
// Kernel C2: combine.  t2[i,j] == t1[j,i]  =>  pairloss = 2*m1.
// ---------------------------------------------------------------------------
__global__ __launch_bounds__(128) void finalize_kernel(
    const float* __restrict__ score, const float* __restrict__ ce_row,
    const float* __restrict__ fns_row, float* __restrict__ out)
{
  __shared__ float red[128];
  const int j = threadIdx.x;
  float total = 0.0f;
  for (int c = 0; c < 3; ++c) {
    float ce  = bsum128(ce_row[c * BB + j],  red) * (1.0f / BB);
    float fns = bsum128(fns_row[c * BB + j], red) * (1.0f / (BB * BB));
    const float* sc = score + (size_t)c * BB * BB;
    float dj = sc[(size_t)j * BB + j];
    float acc = 0.0f;
    for (int i = 0; i < BB; ++i) {
      if (i != j) {
        float di  = sc[(size_t)i * BB + i];
        float sij = sc[(size_t)i * BB + j];
        float sji = sc[(size_t)j * BB + i];
        acc += fabsf(di - sji - dj + sij);   // t1[i,j]
      }
    }
    float m1 = bsum128(acc, red) / (float)(BB * (BB - 1));
    total += ce + 0.1f * fns + 0.2f * m1;    // beta*(relu(m1)+relu(m2)) = 0.2*m1
  }
  if (j == 0) out[0] = total * (1.0f / 3.0f);
}

// ---------------------------------------------------------------------------
extern "C" void kernel_launch(void* const* d_in, const int* in_sizes, int n_in,
                              void* d_out, int out_size, void* d_ws, size_t ws_size,
                              hipStream_t stream) {
  const float* vis = (const float*)d_in[0];   // [B, V, D] f32
  const float* lan = (const float*)d_in[1];   // [B, 3, D] f32
  float* out = (float*)d_out;
  float* ws  = (float*)d_ws;

  // workspace layout (units of 4 bytes); total ~1.58 MB
  _Float16* lan16  = (_Float16*)ws;          //  3*B*D halfs  (98304 slots)
  float*    max0   = ws + 98304;             //  3*B*B
  float*    max1   = ws + 147456;            //  3*B*B
  int*      didx   = (int*)(ws + 196608);    //  3*B (reserve 512)
  float*    lanadj = ws + 197120;            //  3*B*B
  float*    imgadj = ws + 246272;            //  3*B*B
  float*    lgts   = ws + 295424;            //  3*B*B
  float*    score  = ws + 344576;            //  3*B*B
  float*    cerow  = ws + 393728;            //  3*B
  float*    fnsrow = ws + 394112;            //  3*B

  cvt_lan_kernel<<<dim3((3 * BB * DD + 255) / 256), dim3(256), 0, stream>>>(lan, lan16);
  // NOTE: case index is grid.x (fast) so same-a blocks are launch-adjacent.
  sim_topk_kernel<<<dim3(3, BB), dim3(256), 0, stream>>>(vis, lan16, max0, max1, didx);
  adj_kernel<<<dim3(BB, 3), dim3(128), 0, stream>>>(vis, lan, didx, lanadj, imgadj, lgts);
  row_loss_kernel<<<dim3(BB, 3), dim3(128), 0, stream>>>(max0, max1, lanadj, imgadj, lgts,
                                                         score, cerow, fnsrow);
  finalize_kernel<<<dim3(1), dim3(128), 0, stream>>>(score, cerow, fnsrow, out);
}
